// HMM_10264971837538
// MI455X (gfx1250) — matrix operand order repaired and checked by reading, not compiled, not run on hardware
//
#include <hip/hip_runtime.h>
#include <hip/hip_bf16.h>

// CDNA5 / gfx1250, wave32. WMMA f16->f32 16x16x32 from LDS tiles.
// Re-used operands (alpha, T_mask) are pre-converted to f16 once and then
// streamed into LDS with GLOBAL_LOAD_ASYNC_TO_LDS_B128 (ASYNCcnt path);
// per-use exp()-transformed operands stage through VGPRs with v_cvt.

typedef __attribute__((ext_vector_type(16))) _Float16 v16h;
typedef __attribute__((ext_vector_type(8)))  _Float16 v8h;
typedef __attribute__((ext_vector_type(8)))  float    v8f;

#define NEGV (-1e30f)

#if defined(__has_builtin)
#if __has_builtin(__builtin_amdgcn_global_load_async_to_lds_b128)
#define HAVE_ASYNC_LDS 1
#endif
#endif

// Exact pointee type from the builtin's prototype: GCC vector of 4 ints.
typedef int v4i_vs __attribute__((vector_size(4 * sizeof(int))));
typedef __attribute__((address_space(1))) v4i_vs* gptr_b128;
typedef __attribute__((address_space(3))) v4i_vs* lptr_b128;

__device__ __forceinline__ void async_copy16(_Float16* lds_dst, const _Float16* gsrc) {
#ifdef HAVE_ASYNC_LDS
  __builtin_amdgcn_global_load_async_to_lds_b128(
      (gptr_b128)gsrc, (lptr_b128)lds_dst, 0, 0);
#else
  *(v8h*)lds_dst = *(const v8h*)gsrc;
#endif
}

__device__ __forceinline__ void wait_async() {
#ifdef HAVE_ASYNC_LDS
#if __has_builtin(__builtin_amdgcn_s_wait_asynccnt)
  __builtin_amdgcn_s_wait_asynccnt(0);
#else
  asm volatile("s_wait_asynccnt 0" ::: "memory");
#endif
#endif
}

__device__ __forceinline__ float safe_logf(float x) {
  return (x > 0.f) ? __logf(fmaxf(x, 1e-38f)) : NEGV;
}

__device__ __forceinline__ v8f wmma_f16(v16h a, v16h b, v8f c) {
  return __builtin_amdgcn_wmma_f32_16x16x32_f16(
      false, a, false, b, (short)0, c, false, false);
}

__device__ __forceinline__ v8f zero8() {
  v8f z;
#pragma unroll
  for (int i = 0; i < 8; ++i) z[i] = 0.f;
  return z;
}

// A-fragment: 16x32 f16, LDS As[m*32+k] (k contiguous).
__device__ __forceinline__ v16h frag_A(const _Float16* As, int row, int lane) {
  v16h a;
  int kb = (lane & 16) ? 16 : 0;
#pragma unroll
  for (int i = 0; i < 16; ++i) {
    int k = kb + i;
    a[i] = As[row * 32 + k];
  }
  return a;
}

// B-fragment, LDS stored [n][k] (k contiguous, 32 per row).
__device__ __forceinline__ v16h frag_B_kcontig(const _Float16* Bs, int n, int lane) {
  v16h b;
  int kb = (lane & 16) ? 16 : 0;
#pragma unroll
  for (int i = 0; i < 16; ++i) b[i] = Bs[n * 32 + kb + i];
  return b;
}

// B-fragment, LDS stored [k][n] (n contiguous, ldn wide).
__device__ __forceinline__ v16h frag_B_ncontig(const _Float16* Bs, int ldn, int n, int lane) {
  v16h b;
  int kb = (lane & 16) ? 16 : 0;
#pragma unroll
  for (int i = 0; i < 16; ++i) b[i] = Bs[(kb + i) * ldn + n];
  return b;
}

// =============================== reductions ===============================

__global__ __launch_bounds__(256) void k_rowmax(const float* __restrict__ A,
                                                int cols, float* __restrict__ out) {
  int row = blockIdx.x;
  const float* a = A + (size_t)row * cols;
  float m = -3.4e38f;
  for (int c = threadIdx.x; c < cols; c += blockDim.x) m = fmaxf(m, a[c]);
  __shared__ float sm[256];
  sm[threadIdx.x] = m;
  __syncthreads();
  for (int s = 128; s > 0; s >>= 1) {
    if (threadIdx.x < s) sm[threadIdx.x] = fmaxf(sm[threadIdx.x], sm[threadIdx.x + s]);
    __syncthreads();
  }
  if (threadIdx.x == 0) out[row] = sm[0];
}

__global__ __launch_bounds__(256) void k_colmax(const float* __restrict__ B,
                                                const float* __restrict__ Badd,
                                                int rows, size_t cols,
                                                float* __restrict__ out) {
  size_t c = (size_t)blockIdx.x * blockDim.x + threadIdx.x;
  if (c >= cols) return;
  float m = -3.4e38f;
  for (int r = 0; r < rows; ++r) {
    float v = B[(size_t)r * cols + c];
    if (Badd) v += Badd[(size_t)r * cols + c];
    m = fmaxf(m, v);
  }
  out[c] = m;
}

// ============================ f16 pre-conversion ============================

__global__ __launch_bounds__(256) void k_cvt_f16(const float* __restrict__ src,
                                                 _Float16* __restrict__ dst, size_t n) {
  size_t i = (size_t)blockIdx.x * blockDim.x + threadIdx.x;
  if (i < n) dst[i] = (_Float16)src[i];
}

// Pad rows from width V to stride Vp with zeros (alignment + K-tail removal).
__global__ __launch_bounds__(256) void k_cvt_f16_pad(const float* __restrict__ src,
                                                     _Float16* __restrict__ dst,
                                                     int rows, int V, int Vp) {
  size_t i = (size_t)blockIdx.x * blockDim.x + threadIdx.x;
  size_t total = (size_t)rows * Vp;
  if (i >= total) return;
  int r = (int)(i / Vp), v = (int)(i - (size_t)r * Vp);
  dst[i] = (v < V) ? (_Float16)src[(size_t)r * V + v] : (_Float16)0.f;
}

// ======================= scan GEMV helpers (serial) =======================

__global__ __launch_bounds__(1024) void k_prepare_exp(const float* __restrict__ y,
                                                      const float* __restrict__ beta,
                                                      const int* __restrict__ toks,
                                                      int tokIdx, int V, int addBeta,
                                                      float* __restrict__ xvec,
                                                      float* __restrict__ mbuf, int H) {
  __shared__ float sm[1024];
  int tok = addBeta ? toks[tokIdx] : 0;
  float loc = -3.4e38f;
  for (int i = threadIdx.x; i < H; i += blockDim.x) {
    float v = y[i] + (addBeta ? beta[(size_t)i * V + tok] : 0.f);
    loc = fmaxf(loc, v);
  }
  sm[threadIdx.x] = loc;
  __syncthreads();
  for (int s = 512; s > 0; s >>= 1) {
    if (threadIdx.x < s) sm[threadIdx.x] = fmaxf(sm[threadIdx.x], sm[threadIdx.x + s]);
    __syncthreads();
  }
  float m = sm[0];
  for (int i = threadIdx.x; i < H; i += blockDim.x) {
    float v = y[i] + (addBeta ? beta[(size_t)i * V + tok] : 0.f);
    xvec[i] = __expf(v - m);
  }
  if (threadIdx.x == 0) mbuf[0] = m;
}

__global__ __launch_bounds__(256) void k_gemv_col(const float* __restrict__ xvec,
                                                  const float* __restrict__ mbuf,
                                                  const float* __restrict__ alpha,
                                                  float* __restrict__ yout, int H) {
  int j = blockIdx.x * blockDim.x + threadIdx.x;
  if (j >= H) return;
  float acc = 0.f;
  for (int i = 0; i < H; ++i) acc = fmaf(xvec[i], alpha[(size_t)i * H + j], acc);
  yout[j] = safe_logf(acc) + mbuf[0];
}

__global__ __launch_bounds__(256) void k_gemv_row(const float* __restrict__ xvec,
                                                  const float* __restrict__ mbuf,
                                                  const float* __restrict__ alpha,
                                                  const float* __restrict__ beta,
                                                  const int* __restrict__ toks,
                                                  int tokIdx, int V,
                                                  float* __restrict__ yout, int H) {
  int wave = threadIdx.x >> 5, lane = threadIdx.x & 31;
  int j = blockIdx.x * (blockDim.x >> 5) + wave;
  if (j >= H) return;
  const float* row = alpha + (size_t)j * H;
  float acc = 0.f;
  for (int i = lane; i < H; i += 32) acc = fmaf(row[i], xvec[i], acc);
#pragma unroll
  for (int s = 16; s > 0; s >>= 1) acc += __shfl_down(acc, s, 32);
  if (lane == 0) {
    int tok = toks[tokIdx];
    yout[j] = safe_logf(acc) + mbuf[0] + beta[(size_t)j * V + tok];
  }
}

__global__ __launch_bounds__(256) void k_binit(const float* __restrict__ beta,
                                               const int* __restrict__ suffix,
                                               int idx, int V, float* __restrict__ yB, int H) {
  int j = blockIdx.x * blockDim.x + threadIdx.x;
  if (j < H) yB[j] = beta[(size_t)j * V + suffix[idx]];
}

__global__ __launch_bounds__(256) void k_build_ymask(const float* __restrict__ yB,
                                                     const unsigned char* __restrict__ accept,
                                                     float* __restrict__ out, int NSc, int H) {
  int idx = blockIdx.x * blockDim.x + threadIdx.x;
  if (idx >= NSc * H) return;
  int s = idx / H, h = idx - s * H;
  out[idx] = accept[s] ? yB[h] : NEGV;
}

// ============================ WMMA GEMM kernels ============================

// out[m,n] = safe_log( sum_k exp(A[m,k]-rmax[m]) * Bh[n,k] ) + rmax[m]
// Bh = alpha pre-converted to f16 (row-major N x K, k contiguous) -> async LDS.
__global__ __launch_bounds__(256) void k_gemm_rowexp(const float* __restrict__ A,
                                                     const float* __restrict__ rmax,
                                                     const _Float16* __restrict__ Bh,
                                                     float* __restrict__ out,
                                                     int M, int N, int K) {
  __shared__ __attribute__((aligned(16))) _Float16 As[128 * 32];
  __shared__ __attribute__((aligned(16))) _Float16 Bs[64 * 32];
  int tid = threadIdx.x, wave = tid >> 5, lane = tid & 31;
  int m0 = blockIdx.y * 128, n0 = blockIdx.x * 64;
  v8f acc[4];
#pragma unroll
  for (int t = 0; t < 4; ++t) acc[t] = zero8();
  int kIters = K / 32;
  for (int kk = 0; kk < kIters; ++kk) {
    int k0 = kk * 32;
    {  // B tile: async DMA f16 global -> LDS, 16B per lane
      int n = tid >> 2, kb = (tid & 3) * 8;
      async_copy16(&Bs[n * 32 + kb], Bh + (size_t)(n0 + n) * K + k0 + kb);
    }
    {  // A tile: exp-transform through VGPRs (per-use), 16 elems/thread
      int r = tid >> 1, kb = (tid & 1) * 16;
      const float* ap = A + (size_t)(m0 + r) * K + k0 + kb;
      float mr = rmax[m0 + r];
      __builtin_prefetch(ap + 32);
#pragma unroll
      for (int i = 0; i < 16; ++i) As[r * 32 + kb + i] = (_Float16)__expf(ap[i] - mr);
    }
    wait_async();
    __syncthreads();
    v16h af = frag_A(As, wave * 16 + (lane & 15), lane);
#pragma unroll
    for (int t = 0; t < 4; ++t) {
      v16h bf = frag_B_kcontig(Bs, t * 16 + (lane & 15), lane);
      acc[t] = wmma_f16(af, bf, acc[t]);
    }
    __syncthreads();
  }
  int moff = wave * 16 + ((lane & 16) ? 8 : 0);
  int ncol = lane & 15;
#pragma unroll
  for (int t = 0; t < 4; ++t)
#pragma unroll
    for (int r = 0; r < 8; ++r) {
      int m = m0 + moff + r, n = n0 + t * 16 + ncol;
      out[(size_t)m * N + n] = safe_logf(acc[t][r]) + rmax[m];
    }
}

// matmul_a_logb with f16 mask A (async LDS) and B stored [k][n] (n contig):
// out = safe_log( A @ exp(B+Badd - cm[n]) ) + cm[n]
__global__ __launch_bounds__(256) void k_gemm_maskexp(const _Float16* __restrict__ Amh,
                                                      const float* __restrict__ Bsrc,
                                                      const float* __restrict__ Badd,
                                                      const float* __restrict__ cm,
                                                      float* __restrict__ out,
                                                      int M, int N, int K) {
  __shared__ __attribute__((aligned(16))) _Float16 As[128 * 32];
  __shared__ __attribute__((aligned(16))) _Float16 Bs[32 * 64];
  int tid = threadIdx.x, wave = tid >> 5, lane = tid & 31;
  int m0 = blockIdx.y * 128, n0 = blockIdx.x * 64;
  v8f acc[4];
#pragma unroll
  for (int t = 0; t < 4; ++t) acc[t] = zero8();
  int kIters = K / 32;
  for (int kk = 0; kk < kIters; ++kk) {
    int k0 = kk * 32;
    {  // A tile: async DMA f16, two 16B transfers per lane
      int r = tid >> 1, kb = (tid & 1) * 16;
      const _Float16* ap = Amh + (size_t)(m0 + r) * K + k0 + kb;
      async_copy16(&As[r * 32 + kb], ap);
      async_copy16(&As[r * 32 + kb + 8], ap + 8);
    }
    {  // B tile: exp(B+Badd-cm) staged through VGPRs, stored [k][n]
      int k = tid >> 3, nb = (tid & 7) * 8;
      size_t base = (size_t)(k0 + k) * N + n0 + nb;
      __builtin_prefetch(Bsrc + base + (size_t)32 * N);
#pragma unroll
      for (int i = 0; i < 8; ++i) {
        int n = n0 + nb + i;
        float v = 0.f;
        if (n < N) {
          v = Bsrc[base + i];
          if (Badd) v += Badd[base + i];
          v = __expf(v - cm[n]);
        }
        Bs[k * 64 + nb + i] = (_Float16)v;
      }
    }
    wait_async();
    __syncthreads();
    v16h af = frag_A(As, wave * 16 + (lane & 15), lane);
#pragma unroll
    for (int t = 0; t < 4; ++t) {
      v16h bf = frag_B_ncontig(Bs, 64, t * 16 + (lane & 15), lane);
      acc[t] = wmma_f16(af, bf, acc[t]);
    }
    __syncthreads();
  }
  int moff = wave * 16 + ((lane & 16) ? 8 : 0);
  int ncol = lane & 15;
#pragma unroll
  for (int t = 0; t < 4; ++t)
#pragma unroll
    for (int r = 0; r < 8; ++r) {
      int m = m0 + moff + r, n = n0 + t * 16 + ncol;
      if (n < N) out[(size_t)m * N + n] = safe_logf(acc[t][r]) + cm[n];
    }
}

// Raw f16 x f16 GEMM (mask product): out = A @ B, A (MxK, async), B [k][n]
// taken from padded TmaskH (row stride Vp, async). Stores guarded to n < N.
__global__ __launch_bounds__(256) void k_gemm_maskraw(const _Float16* __restrict__ Amh,
                                                      const _Float16* __restrict__ Bh,
                                                      float* __restrict__ out,
                                                      int M, int N, int Vp, int K) {
  __shared__ __attribute__((aligned(16))) _Float16 As[128 * 32];
  __shared__ __attribute__((aligned(16))) _Float16 Bs[32 * 64];
  int tid = threadIdx.x, wave = tid >> 5, lane = tid & 31;
  int m0 = blockIdx.y * 128, n0 = blockIdx.x * 64;
  v8f acc[4];
#pragma unroll
  for (int t = 0; t < 4; ++t) acc[t] = zero8();
  int kIters = K / 32;
  for (int kk = 0; kk < kIters; ++kk) {
    int k0 = kk * 32;
    {  // A tile async
      int r = tid >> 1, kb = (tid & 1) * 16;
      const _Float16* ap = Amh + (size_t)(m0 + r) * K + k0 + kb;
      async_copy16(&As[r * 32 + kb], ap);
      async_copy16(&As[r * 32 + kb + 8], ap + 8);
    }
    {  // B tile async: [k][n], 8 halves per lane
      int k = tid >> 3, nb = (tid & 7) * 8;
      async_copy16(&Bs[k * 64 + nb], Bh + (size_t)(k0 + k) * Vp + n0 + nb);
    }
    wait_async();
    __syncthreads();
    v16h af = frag_A(As, wave * 16 + (lane & 15), lane);
#pragma unroll
    for (int t = 0; t < 4; ++t) {
      v16h bf = frag_B_ncontig(Bs, 64, t * 16 + (lane & 15), lane);
      acc[t] = wmma_f16(af, bf, acc[t]);
    }
    __syncthreads();
  }
  int moff = wave * 16 + ((lane & 16) ? 8 : 0);
  int ncol = lane & 15;
#pragma unroll
  for (int t = 0; t < 4; ++t)
#pragma unroll
    for (int r = 0; r < 8; ++r) {
      int m = m0 + moff + r, n = n0 + t * 16 + ncol;
      if (n < N) out[(size_t)m * N + n] = acc[t][r];
    }
}

// T_weights[e,h] = safe_log( sum_v TmaskH[e,v]*exp(beta[h,v]-bmaxRow[h]) ) + bmaxRow[h]
// A = padded f16 T_mask (stride Vp, async); B = exp(beta row) staged (k contig).
__global__ __launch_bounds__(256) void k_gemm_Tw(const _Float16* __restrict__ TmaskH,
                                                 const float* __restrict__ beta,
                                                 const float* __restrict__ bmaxRow,
                                                 float* __restrict__ Tw,
                                                 int E_, int Hn, int V_, int Vp) {
  __shared__ __attribute__((aligned(16))) _Float16 As[128 * 32];
  __shared__ __attribute__((aligned(16))) _Float16 Bs[64 * 32];
  int tid = threadIdx.x, wave = tid >> 5, lane = tid & 31;
  int m0 = blockIdx.y * 128, n0 = blockIdx.x * 64;  // m0: e, n0: h
  v8f acc[4];
#pragma unroll
  for (int t = 0; t < 4; ++t) acc[t] = zero8();
  int kIters = Vp / 32;  // Vp = 32*ceil(V/32): zero-padded, no A tail checks
  for (int kk = 0; kk < kIters; ++kk) {
    int k0 = kk * 32;
    {  // A tile async (padded f16 mask)
      int r = tid >> 1, kb = (tid & 1) * 16;
      const _Float16* ap = TmaskH + (size_t)(m0 + r) * Vp + k0 + kb;
      async_copy16(&As[r * 32 + kb], ap);
      async_copy16(&As[r * 32 + kb + 8], ap + 8);
    }
    {  // B tile: exp(beta[h,v]-bmax[h]) staged, zero-pad v tail
      int n = tid >> 2, kb = (tid & 3) * 8;
      const float* bp = beta + (size_t)(n0 + n) * V_ + k0 + kb;
      float mr = bmaxRow[n0 + n];
      __builtin_prefetch(bp + 32);
#pragma unroll
      for (int i = 0; i < 8; ++i) {
        int k = k0 + kb + i;
        Bs[n * 32 + kb + i] = (k < V_) ? (_Float16)__expf(bp[i] - mr) : (_Float16)0.f;
      }
    }
    wait_async();
    __syncthreads();
    v16h af = frag_A(As, wave * 16 + (lane & 15), lane);
#pragma unroll
    for (int t = 0; t < 4; ++t) {
      v16h bf = frag_B_kcontig(Bs, t * 16 + (lane & 15), lane);
      acc[t] = wmma_f16(af, bf, acc[t]);
    }
    __syncthreads();
  }
  int moff = wave * 16 + ((lane & 16) ? 8 : 0);
  int ncol = lane & 15;
#pragma unroll
  for (int t = 0; t < 4; ++t)
#pragma unroll
    for (int r = 0; r < 8; ++r) {
      int m = m0 + moff + r, n = n0 + t * 16 + ncol;
      Tw[(size_t)m * Hn + n] = safe_logf(acc[t][r]) + bmaxRow[n];
    }
}

// matmul_log: ls[m,v] = safe_log(sum_h exp(Z[m,h]-rmax[m])*exp(beta[h,v]-cmax[v]))
//                       + rmax[m] + cmax[v]. Tile 16(M) x 128(N), 8 waves.
__global__ __launch_bounds__(256) void k_gemm_logits(const float* __restrict__ Z,
                                                     const float* __restrict__ rmax,
                                                     const float* __restrict__ beta,
                                                     const float* __restrict__ cmax,
                                                     float* __restrict__ ls,
                                                     int Mext, int V_, int Hn) {
  __shared__ __attribute__((aligned(16))) _Float16 As[16 * 32];
  __shared__ __attribute__((aligned(16))) _Float16 Bs[32 * 128];
  int tid = threadIdx.x, wave = tid >> 5, lane = tid & 31;
  int m0 = blockIdx.y * 16, n0 = blockIdx.x * 128;
  v8f acc = zero8();
  int kIters = Hn / 32;
  for (int kk = 0; kk < kIters; ++kk) {
    int k0 = kk * 32;
    {
#pragma unroll
      for (int i = 0; i < 2; ++i) {
        int e = tid * 2 + i;
        int r = e >> 5, k = e & 31;
        float mr = rmax[m0 + r];
        As[e] = (_Float16)__expf(Z[(size_t)(m0 + r) * Hn + k0 + k] - mr);
      }
    }
    {
      int k = tid >> 3, nb = (tid & 7) * 16;
      const float* bp = beta + (size_t)(k0 + k) * V_ + n0 + nb;
      __builtin_prefetch(bp + (size_t)32 * V_);
#pragma unroll
      for (int i = 0; i < 16; ++i) {
        int n = n0 + nb + i;
        float v = (n < V_) ? __expf(bp[i] - cmax[n]) : 0.f;
        Bs[k * 128 + nb + i] = (_Float16)v;
      }
    }
    __syncthreads();
    v16h af = frag_A(As, (lane & 15), lane);
    v16h bf = frag_B_ncontig(Bs, 128, wave * 16 + (lane & 15), lane);
    acc = wmma_f16(af, bf, acc);
    __syncthreads();
  }
  int moff = (lane & 16) ? 8 : 0;
  int ncol = lane & 15;
#pragma unroll
  for (int r = 0; r < 8; ++r) {
    int m = m0 + moff + r, n = n0 + wave * 16 + ncol;
    if (n < V_) ls[(size_t)m * V_ + n] = safe_logf(acc[r]) + rmax[m] + cmax[n];
  }
}

// ============================ misc epilogues ============================

__global__ __launch_bounds__(256) void k_cranges(const float* __restrict__ Cbuf,
                                                 const float* __restrict__ cm3,
                                                 const float* __restrict__ rmask,
                                                 float* __restrict__ out,
                                                 int Tp1, size_t NSH) {
  __shared__ float rm[16 * 65];
  for (int i = threadIdx.x; i < 16 * Tp1; i += blockDim.x) rm[i] = rmask[i];
  __syncthreads();
  size_t col = (size_t)blockIdx.x * blockDim.x + threadIdx.x;
  if (col >= NSH) return;
  float m = cm3[col];
  float acc[16];
#pragma unroll
  for (int r = 0; r < 16; ++r) acc[r] = 0.f;
  for (int t = 0; t < Tp1; ++t) {
    float v = __expf(Cbuf[(size_t)t * NSH + col] - m);
#pragma unroll
    for (int r = 0; r < 16; ++r) acc[r] = fmaf(rm[r * Tp1 + t], v, acc[r]);
  }
#pragma unroll
  for (int r = 0; r < 16; ++r) out[(size_t)r * NSH + col] = safe_logf(acc[r]) + m;
}

__global__ __launch_bounds__(256) void k_build_Z(const float* __restrict__ yA,
                                                 const float* __restrict__ Cr0,
                                                 float* __restrict__ Z,
                                                 int Hn, int NSc, int Mext) {
  int idx = blockIdx.x * blockDim.x + threadIdx.x;
  if (idx >= Mext * Hn) return;
  int m = idx / Hn, h = idx - m * Hn;
  float v;
  if (m < NSc) v = yA[h] + Cr0[(size_t)m * Hn + h];
  else if (m == NSc) v = yA[h];
  else v = NEGV;
  Z[idx] = v;
}

__global__ __launch_bounds__(256) void k_build_maskA(const float* __restrict__ VE,
                                                     const float* __restrict__ EV,
                                                     const int* __restrict__ dstate,
                                                     _Float16* __restrict__ maskAh,
                                                     int NSc, int E_) {
  int idx = blockIdx.x * blockDim.x + threadIdx.x;
  if (idx >= NSc * E_) return;
  int s = idx / E_, e = idx - s * E_;
  int d = dstate[0];
  maskAh[idx] = (_Float16)(VE[(size_t)d * E_ + e] * EV[(size_t)e * NSc + s]);
}

__global__ __launch_bounds__(256) void k_final(const float* __restrict__ ls,
                                               const float* __restrict__ maskM,
                                               float* __restrict__ out, int V_, int NSc) {
  int v = blockIdx.x * blockDim.x + threadIdx.x;
  if (v >= V_) return;
  float m = -3.4e38f, ssum = 0.f;
  for (int s = 0; s < NSc; ++s) {
    float lv = ls[(size_t)s * V_ + v] + safe_logf(maskM[(size_t)s * V_ + v]);
    if (lv > m) { ssum = ssum * __expf(m - lv) + 1.f; m = lv; }
    else        { ssum += __expf(lv - m); }
  }
  out[v] = m + __logf(ssum);
  out[(size_t)V_ + v] = ls[(size_t)NSc * V_ + v];
}

// ================================ driver ================================

extern "C" void kernel_launch(void* const* d_in, const int* in_sizes, int n_in,
                              void* d_out, int out_size, void* d_ws, size_t ws_size,
                              hipStream_t stream) {
  const int H = 2048, V = 50257, NS = 128, E = 512, T = 64, P = 32, S = 16;
  const int Vp = ((V + 31) / 32) * 32;  // 50272: 16B-aligned f16 row stride
  const size_t NSH = (size_t)NS * H;
  const int Mext = 160;

  const float* alpha  = (const float*)d_in[0];
  const float* beta   = (const float*)d_in[1];
  const float* gamma  = (const float*)d_in[2];
  const float* VE     = (const float*)d_in[3];
  const float* EV     = (const float*)d_in[4];
  const float* Tmask  = (const float*)d_in[5];
  const float* rmask  = (const float*)d_in[6];
  const unsigned char* accept = (const unsigned char*)d_in[7];
  const int* prefix   = (const int*)d_in[8];
  const int* suffix   = (const int*)d_in[9];
  const int* dstate   = (const int*)d_in[10];
  float* out = (float*)d_out;
  (void)in_sizes; (void)n_in; (void)out_size; (void)ws_size;

  size_t off = 0;
  auto allocB = [&](size_t bytes) -> void* {
    void* p = (char*)d_ws + off;
    off += ((bytes + 255) / 256) * 256;
    return p;
  };
  auto allocF = [&](size_t n) -> float* { return (float*)allocB(n * 4); };
  auto allocH = [&](size_t n) -> _Float16* { return (_Float16*)allocB(n * 2); };

  float* bmaxRow = allocF(H);
  float* cmaxCol = allocF(V);
  float* xvec    = allocF(H);
  float* mbuf    = allocF(16);
  float* yA      = allocF(H);
  float* yA2     = allocF(H);
  float* yB      = allocF(H);
  float* yB2     = allocF(H);
  float* Tw      = allocF((size_t)E * H);
  float* Cbuf    = allocF((size_t)(T + 1) * NSH);
  float* y1buf   = allocF((size_t)E * H);
  float* cmH     = allocF(H);
  float* rmaxBuf = allocF(256);
  float* yTmp    = allocF(NSH);
  float* cm3     = allocF(NSH);
  float* Zext    = allocF((size_t)Mext * H);
  float* ls      = allocF((size_t)Mext * V);
  float* maskM   = allocF((size_t)NS * V);
  _Float16* alphaH = allocH((size_t)H * H);
  _Float16* TmaskH = allocH((size_t)E * Vp + 64);  // slack for n-tail overreach
  _Float16* EVh    = allocH((size_t)E * NS);
  _Float16* VEh    = allocH((size_t)NS * E);
  _Float16* maskAh = allocH((size_t)NS * E);

  // 0) f16 pre-conversions of re-used operands
  {
    size_t nA = (size_t)H * H;
    k_cvt_f16<<<(int)((nA + 255) / 256), 256, 0, stream>>>(alpha, alphaH, nA);
    size_t nT = (size_t)E * Vp;
    k_cvt_f16_pad<<<(int)((nT + 255) / 256), 256, 0, stream>>>(Tmask, TmaskH, E, V, Vp);
    size_t nE = (size_t)E * NS;
    k_cvt_f16<<<(int)((nE + 255) / 256), 256, 0, stream>>>(EV, EVh, nE);
    k_cvt_f16<<<(int)((nE + 255) / 256), 256, 0, stream>>>(VE, VEh, nE);
  }

  // 1) beta row/col maxes
  k_rowmax<<<H, 256, 0, stream>>>(beta, V, bmaxRow);
  k_colmax<<<(V + 255) / 256, 256, 0, stream>>>(beta, nullptr, H, (size_t)V, cmaxCol);

  // 2) prefix scan: A vector
  (void)hipMemcpyAsync(yA, gamma, H * sizeof(float), hipMemcpyDeviceToDevice, stream);
  float* pa = yA; float* pb = yA2;
  for (int p = 0; p < P; ++p) {
    k_prepare_exp<<<1, 1024, 0, stream>>>(pa, beta, prefix, p, V, 1, xvec, mbuf, H);
    k_gemv_col<<<H / 256, 256, 0, stream>>>(xvec, mbuf, alpha, pb, H);
    float* t = pa; pa = pb; pb = t;
  }
  float* Avec = pa;

  // 3) suffix scan: B0 vector
  k_binit<<<H / 256, 256, 0, stream>>>(beta, suffix, S - 1, V, yB, H);
  float* qa = yB; float* qb = yB2;
  for (int i = 0; i < S - 1; ++i) {
    k_prepare_exp<<<1, 1024, 0, stream>>>(qa, beta, nullptr, 0, V, 0, xvec, mbuf, H);
    k_gemv_row<<<H / 8, 256, 0, stream>>>(xvec, mbuf, alpha, beta, suffix, S - 2 - i, V, qb, H);
    float* t = qa; qa = qb; qb = t;
  }
  float* Bvec = qa;

  // 4) T_weights (K = Vp, zero-padded)
  {
    dim3 g(H / 64, E / 128);
    k_gemm_Tw<<<g, 256, 0, stream>>>(TmaskH, beta, bmaxRow, Tw, E, H, V, Vp);
  }

  // 5) y0 -> C[0]
  k_build_ymask<<<(NS * H + 255) / 256, 256, 0, stream>>>(Bvec, accept, yTmp, NS, H);
  k_rowmax<<<NS, 256, 0, stream>>>(yTmp, H, rmaxBuf);
  dim3 gRE(H / 64, NS / 128);
  k_gemm_rowexp<<<gRE, 256, 0, stream>>>(yTmp, rmaxBuf, alphaH, Cbuf, NS, H, H);

  // 6) c_step x T
  for (int t = 0; t < T; ++t) {
    const float* yc = Cbuf + (size_t)t * NSH;
    k_colmax<<<(H + 255) / 256, 256, 0, stream>>>(yc, nullptr, NS, (size_t)H, cmH);
    dim3 g1(H / 64, E / 128);
    k_gemm_maskexp<<<g1, 256, 0, stream>>>(EVh, yc, nullptr, cmH, y1buf, E, H, NS);
    k_colmax<<<(H + 255) / 256, 256, 0, stream>>>(y1buf, Tw, E, (size_t)H, cmH);
    dim3 g2(H / 64, NS / 128);
    k_gemm_maskexp<<<g2, 256, 0, stream>>>(VEh, y1buf, Tw, cmH, yTmp, NS, H, E);
    k_rowmax<<<NS, 256, 0, stream>>>(yTmp, H, rmaxBuf);
    k_gemm_rowexp<<<gRE, 256, 0, stream>>>(yTmp, rmaxBuf, alphaH,
                                           Cbuf + (size_t)(t + 1) * NSH, NS, H, H);
  }

  // 7) C_ranges -> d_out[2V ..]
  float* outCr = out + 2 * (size_t)V;
  k_colmax<<<(int)((NSH + 255) / 256), 256, 0, stream>>>(Cbuf, nullptr, T + 1, NSH, cm3);
  k_cranges<<<(int)((NSH + 255) / 256), 256, 0, stream>>>(Cbuf, cm3, rmask, outCr, T + 1, NSH);

  // 8) Z and logits_states
  k_build_Z<<<(Mext * H + 255) / 256, 256, 0, stream>>>(Avec, outCr, Zext, H, NS, Mext);
  k_rowmax<<<Mext, 256, 0, stream>>>(Zext, H, rmaxBuf);
  {
    dim3 g((V + 127) / 128, Mext / 16);
    k_gemm_logits<<<g, 256, 0, stream>>>(Zext, rmaxBuf, beta, cmaxCol, ls, Mext, V, H);
  }

  // 9) mask product (all-f16 async WMMA) + masked logsumexp
  k_build_maskA<<<(NS * E + 255) / 256, 256, 0, stream>>>(VE, EV, dstate, maskAh, NS, E);
  {
    dim3 g((V + 63) / 64, NS / 128);
    k_gemm_maskraw<<<g, 256, 0, stream>>>(maskAh, TmaskH, maskM, NS, V, Vp, E);
  }
  k_final<<<(V + 255) / 256, 256, 0, stream>>>(ls, maskM, out, V, NS);
}
